// SAWindowCALayer_42812234006556
// MI455X (gfx1250) — compile-verified, matrix operand order checked by reading, not run on hardware
//
#include <hip/hip_runtime.h>
#include <cmath>

// ---------------------------------------------------------------------------
// SAWindowCALayer for MI455X (gfx1250, wave32, WMMA).
// Heavy GEMMs: v_wmma_f32_16x16x32_bf16 (bf16 in, f32 accumulate).
// Operand staging: b128 global loads -> packed b128 LDS stores; WMMA operands
// assembled with two ds_load_b128 each (K={kb..kb+7},{kb+16..kb+23} contiguous
// per lane in the CDNA5 16-bit operand layout).
// Workspace requirement: 16,777,216 f32 + 67,108,864 bf16  ==  ~201 MB.
// ---------------------------------------------------------------------------

typedef __bf16 bf16;
typedef __attribute__((ext_vector_type(16))) __bf16 v16bf;
typedef __attribute__((ext_vector_type(8)))  __bf16 v8bf;
typedef __attribute__((ext_vector_type(8)))  float  v8f;

// K-index held by vector element e on lane `lane` for the 16-bit 16x32 A/B
// WMMA operand layout (CDNA5 ISA §7.12.2):
//   lanes 0-15 : K = {0..7, 16..23},  lanes 16-31 : K = {8..15, 24..31}
__device__ __forceinline__ int kmap(int e, int lane) {
  return e + (e & 8) + ((lane & 16) >> 1);
}

// Build one 16x32 bf16 WMMA operand from an LDS row whose leading dimension is
// the operand's M (or N) index: two 16-byte ds_load_b128, no repacking.
__device__ __forceinline__ v16bf load_frag(const bf16* row_ptr, int kb) {
  v8bf lo = *(const v8bf*)(row_ptr + kb);        // K = kb .. kb+7
  v8bf hi = *(const v8bf*)(row_ptr + kb + 16);   // K = kb+16 .. kb+23
  return __builtin_shufflevector(lo, hi, 0, 1, 2, 3, 4, 5, 6, 7,
                                         8, 9, 10, 11, 12, 13, 14, 15);
}

__device__ __forceinline__ float gelu_exact(float x) {
  return 0.5f * x * (1.0f + erff(x * 0.70710678118654752f));
}

__device__ __forceinline__ float block_reduce_sum256(float v, float* sb, int t) {
  sb[t] = v; __syncthreads();
  for (int o = 128; o > 0; o >>= 1) { if (t < o) sb[t] += sb[t + o]; __syncthreads(); }
  float r = sb[0]; __syncthreads();
  return r;
}

// ---------------------------------------------------------------------------
// Generic GEMM: out = act(A[MxK] @ W[KxN] + bias + res), bf16 WMMA, f32 acc.
// Block tile 32x64 (8 waves of 16x16), K staged in chunks of 64 (2 WMMA per
// barrier pair). Requires M%32==0, N%64==0, K%64==0 (true for every call).
// ---------------------------------------------------------------------------
__global__ __launch_bounds__(256) void gemm_bf16_wmma(
    const float* __restrict__ A, const float* __restrict__ W,
    const float* __restrict__ bias, const float* __restrict__ res,
    float* __restrict__ outF, bf16* __restrict__ outB,
    int M, int N, int K, int act)
{
  // 72-half row stride: 144 B = multiple of 16 B (aligned b128) and 36 banks
  // (16 lanes -> 16 distinct bank quads, conflict-free fragment loads).
  __shared__ __align__(16) bf16 As[32][72];     // [m][k]
  __shared__ __align__(16) bf16 Bst[64][72];    // [n][k]  (transposed W tile)
  const int tid  = threadIdx.x;
  const int lane = tid & 31;
  const int wv   = tid >> 5;
  const int wm   = wv >> 2;         // 0..1 : 16-row slice
  const int wn   = wv & 3;          // 0..3 : 16-col slice
  const int m0   = blockIdx.y * 32;
  const int n0   = blockIdx.x * 64;
  const int kb   = (lane & 16) >> 1;

  const int ar = tid >> 3;          // A stage: row 0..31
  const int ac = (tid & 7) * 8;     //          8 cols per thread
  const int bc = (tid & 15) * 4;    // B stage: 4 cols per thread
  const int br = tid >> 4;          //          row group 0..15

  v8f acc = {};
  for (int k0 = 0; k0 < K; k0 += 64) {
    // ---- stage A tile 32x64 (f32 -> bf16), one b128 LDS store per thread ----
    {
      const float4 a4a = *(const float4*)&A[(size_t)(m0 + ar) * K + k0 + ac];
      const float4 a4b = *(const float4*)&A[(size_t)(m0 + ar) * K + k0 + ac + 4];
      v8bf v;
      v[0] = (bf16)a4a.x; v[1] = (bf16)a4a.y; v[2] = (bf16)a4a.z; v[3] = (bf16)a4a.w;
      v[4] = (bf16)a4b.x; v[5] = (bf16)a4b.y; v[6] = (bf16)a4b.z; v[7] = (bf16)a4b.w;
      *(v8bf*)&As[ar][ac] = v;
    }
    // ---- stage W tile 64x64 transposed into Bst[n][k] ----
#pragma unroll
    for (int g = 0; g < 4; ++g) {
      int kr = br * 4 + g;
      const float4 w4 = *(const float4*)&W[(size_t)(k0 + kr) * N + n0 + bc];
      Bst[bc + 0][kr] = (bf16)w4.x;
      Bst[bc + 1][kr] = (bf16)w4.y;
      Bst[bc + 2][kr] = (bf16)w4.z;
      Bst[bc + 3][kr] = (bf16)w4.w;
    }
    if (k0 + 64 < K) {  // lowers to global_prefetch_b8: warm next K-tile
      __builtin_prefetch(&A[(size_t)(m0 + ar) * K + k0 + 64], 0, 1);
      __builtin_prefetch(&W[(size_t)(k0 + 64 + (tid >> 2)) * N + n0], 0, 1);
    }
    __syncthreads();

    const bf16* arow = &As[wm * 16 + (lane & 15)][0];
    const bf16* brow = &Bst[wn * 16 + (lane & 15)][0];
    v16bf a0 = load_frag(arow, kb);
    v16bf b0 = load_frag(brow, kb);
    acc = __builtin_amdgcn_wmma_f32_16x16x32_bf16(false, a0, false, b0, (short)0, acc, false, false);
    v16bf a1 = load_frag(arow + 32, kb);
    v16bf b1 = load_frag(brow + 32, kb);
    acc = __builtin_amdgcn_wmma_f32_16x16x32_bf16(false, a1, false, b1, (short)0, acc, false, false);
    __syncthreads();
  }

#pragma unroll
  for (int r = 0; r < 8; ++r) {
    int cm = r + ((lane & 16) >> 1);    // C/D layout: lanes>=16 hold rows 8..15
    int cn = lane & 15;
    size_t row = (size_t)(m0 + wm * 16 + cm);
    size_t col = (size_t)(n0 + wn * 16 + cn);
    float v = acc[r];
    if (bias) v += bias[col];
    if (res)  v += res[row * N + col];
    if (act)  v = gelu_exact(v);
    if (outF) outF[row * N + col] = v;
    else      outB[row * N + col] = (bf16)v;
  }
}

// ---------------------------------------------------------------------------
// Causal self-attention with fused RoPE, one block per (b, h, 16-query tile).
// Scores/probabilities live in a 16x1028 f32 LDS buffer (padded stride: the
// P@V fragment build reads 16 rows in parallel, bank = 4*m, conflict-free);
// QK^T and P@V both run on WMMA.
// ---------------------------------------------------------------------------
__global__ __launch_bounds__(256) void flash_sa_rope(
    const float* __restrict__ qkv, float* __restrict__ outp)
{
  __shared__ float sc[16][1028];                      // scores -> exp(scores)
  __shared__ __align__(16) bf16 qt[16][72];           // RoPE'd Q tile
  __shared__ __align__(16) bf16 ktile[8][16][72];     // per-wave RoPE'd K tiles
  __shared__ float red[16][16];
  __shared__ float rowmax[16];
  __shared__ float rowsum[16];

  const int qt_i = blockIdx.x;   // 0..63 query tile
  const int h    = blockIdx.y;   // 0..7
  const int b    = blockIdx.z;   // 0..1
  const int tid  = threadIdx.x;
  const int lane = tid & 31;
  const int wv   = tid >> 5;
  const int kb   = (lane & 16) >> 1;

  // Phase 0: load 16x64 Q tile, apply RoPE, pack 8 bf16 per b128 store.
  if (tid < 128) {
    int m = tid >> 3, d0 = (tid & 7) * 8;
    int s = qt_i * 16 + m;
    size_t base = ((size_t)(b * 1024 + s)) * 1536 + h * 64;
    v8bf q8;
#pragma unroll
    for (int i = 0; i < 8; ++i) {
      int d = d0 + i;
      float theta = __powf(10000.0f, -(float)(2 * (d & 31)) * (1.0f / 64.0f));
      float fr = (float)s * theta;
      float qv = qkv[base + d];
      float qp = (d < 32) ? -qkv[base + d + 32] : qkv[base + d - 32];
      q8[i] = (bf16)(qv * __cosf(fr) + qp * __sinf(fr));
    }
    *(v8bf*)&qt[m][d0] = q8;
  }
  __syncthreads();

  // Phase 1: causal score tiles, waves stripe over key tiles kt <= qt_i.
  for (int kt = wv; kt <= qt_i; kt += 8) {
    int r = lane >> 1;                       // key row within tile
    int kpos = kt * 16 + r;
    size_t kbase = ((size_t)(b * 1024 + kpos)) * 1536 + 512 + h * 64;
    int d0 = (lane & 1) * 32;
#pragma unroll
    for (int j = 0; j < 4; ++j) {
      v8bf k8;
#pragma unroll
      for (int i = 0; i < 8; ++i) {
        int d = d0 + j * 8 + i;
        float theta = __powf(10000.0f, -(float)(2 * (d & 31)) * (1.0f / 64.0f));
        float fr = (float)kpos * theta;
        float kv = qkv[kbase + d];
        float kp = (d < 32) ? -qkv[kbase + d + 32] : qkv[kbase + d - 32];
        k8[i] = (bf16)(kv * __cosf(fr) + kp * __sinf(fr));
      }
      *(v8bf*)&ktile[wv][r][d0 + j * 8] = k8;
    }
    const bf16* arow = &qt[lane & 15][0];
    const bf16* brow = &ktile[wv][lane & 15][0];   // B[k=dh][n=key] = ktile[key][dh]
    v16bf a0 = load_frag(arow, kb);
    v16bf b0 = load_frag(brow, kb);
    v8f c = {};
    c = __builtin_amdgcn_wmma_f32_16x16x32_bf16(false, a0, false, b0, (short)0, c, false, false);
    v16bf a1 = load_frag(arow + 32, kb);
    v16bf b1 = load_frag(brow + 32, kb);
    c = __builtin_amdgcn_wmma_f32_16x16x32_bf16(false, a1, false, b1, (short)0, c, false, false);
#pragma unroll
    for (int rr = 0; rr < 8; ++rr) {
      int m = rr + ((lane & 16) >> 1);
      int n = lane & 15;
      int srow = qt_i * 16 + m;
      int col  = kt * 16 + n;
      sc[m][col] = (col <= srow) ? c[rr] * 0.125f : -1.0e30f;
    }
  }
  __syncthreads();

  // Phase 2: row softmax (16 threads per row).
  {
    int row = tid >> 4, sub = tid & 15;
    int srow = qt_i * 16 + row, len = srow + 1;
    float mx = -3.0e38f;
    for (int j = sub; j < len; j += 16) mx = fmaxf(mx, sc[row][j]);
    red[row][sub] = mx; __syncthreads();
    if (sub == 0) {
      float m2 = red[row][0];
      for (int i = 1; i < 16; ++i) m2 = fmaxf(m2, red[row][i]);
      rowmax[row] = m2;
    }
    __syncthreads();
    mx = rowmax[row];
    float sm = 0.0f;
    for (int j = sub; j < len; j += 16) {
      float e = __expf(sc[row][j] - mx);
      sc[row][j] = e;
      sm += e;
    }
    red[row][sub] = sm; __syncthreads();
    if (sub == 0) {
      float s2 = 0.0f;
      for (int i = 0; i < 16; ++i) s2 += red[row][i];
      rowsum[row] = s2;
    }
    __syncthreads();
  }

  // Phase 3: out = P @ V. Waves 0..3 own the 4 dh n-tiles (wave-uniform
  // branch, EXEC stays all-ones inside).
  if (wv < 4) {
    int nt = wv;
    int nchunk = (qt_i * 16 + 16 + 31) >> 5;
    int mrow = lane & 15;
    int srow = qt_i * 16 + mrow;
    float inv = 1.0f / rowsum[mrow];
    v8f c = {};
    for (int ch = 0; ch < nchunk; ++ch) {
      v16bf a, bb;
#pragma unroll
      for (int e = 0; e < 16; ++e) {
        int kk = kmap(e, lane);
        int key = ch * 32 + kk;
        float p = (key <= srow) ? sc[mrow][key] * inv : 0.0f;
        a[e]  = (bf16)p;
        bb[e] = (bf16)qkv[((size_t)(b * 1024 + key)) * 1536 + 1024 + h * 64
                          + nt * 16 + (lane & 15)];
      }
      c = __builtin_amdgcn_wmma_f32_16x16x32_bf16(false, a, false, bb, (short)0, c, false, false);
    }
#pragma unroll
    for (int rr = 0; rr < 8; ++rr) {
      int m = rr + ((lane & 16) >> 1);
      int n = lane & 15;
      outp[((size_t)(b * 1024 + qt_i * 16 + m)) * 512 + h * 64 + nt * 16 + n] = c[rr];
    }
  }
}

// ---------------------------------------------------------------------------
// LayerNorm (D=512), one block per row, optional fused residual.
// ---------------------------------------------------------------------------
__global__ __launch_bounds__(256) void ln_kernel(
    const float* __restrict__ x, const float* __restrict__ res,
    const float* __restrict__ g, const float* __restrict__ be,
    float* __restrict__ y)
{
  __shared__ float sb[256];
  int row = blockIdx.x, t = threadIdx.x;
  size_t base = (size_t)row * 512;
  float v0 = x[base + t]       + (res ? res[base + t]       : 0.0f);
  float v1 = x[base + 256 + t] + (res ? res[base + 256 + t] : 0.0f);
  float mu = block_reduce_sum256(v0 + v1, sb, t) * (1.0f / 512.0f);
  float d0 = v0 - mu, d1 = v1 - mu;
  float var = block_reduce_sum256(d0 * d0 + d1 * d1, sb, t) * (1.0f / 512.0f);
  float rs = rsqrtf(var + 1e-5f);
  y[base + t]       = d0 * rs * g[t]       + be[t];
  y[base + 256 + t] = d1 * rs * g[t + 256] + be[t + 256];
}

// ---------------------------------------------------------------------------
// Ducking gate + LayerNorm, one block per row.
// ---------------------------------------------------------------------------
__global__ __launch_bounds__(256) void gate_ln(
    const float* __restrict__ sar, const float* __restrict__ ca,
    const float* __restrict__ ctmp, const float* __restrict__ g,
    const float* __restrict__ be, float* __restrict__ y)
{
  __shared__ float sb[256];
  int row = blockIdx.x, t = threadIdx.x;
  size_t base = (size_t)row * 512;
  float a0 = sar[base + t], a1 = sar[base + 256 + t];
  float c0 = ca[base + t],  c1 = ca[base + 256 + t];
  float dt = block_reduce_sum256(a0 * c0 + a1 * c1, sb, t);
  float na = block_reduce_sum256(a0 * a0 + a1 * a1, sb, t);
  float nb = block_reduce_sum256(c0 * c0 + c1 * c1, sb, t);
  float cosv = dt / (fmaxf(sqrtf(na), 1e-8f) * fmaxf(sqrtf(nb), 1e-8f));
  float gate = 1.0f / (1.0f + __expf(-ctmp[0] * cosv));
  float x0 = a0 * gate + c0, x1 = a1 * gate + c1;
  float mu = block_reduce_sum256(x0 + x1, sb, t) * (1.0f / 512.0f);
  float d0 = x0 - mu, d1 = x1 - mu;
  float var = block_reduce_sum256(d0 * d0 + d1 * d1, sb, t) * (1.0f / 512.0f);
  float rs = rsqrtf(var + 1e-5f);
  y[base + t]       = d0 * rs * g[t]       + be[t];
  y[base + 256 + t] = d1 * rs * g[t + 256] + be[t + 256];
}

// ---------------------------------------------------------------------------
// Window cross-attention over level 0 (T=512, F=64, start=0), one block per
// (b, s), 128 threads = 128 window slots. K/V gathers hit L2 (134 MB bf16
// resident in the 192 MB L2). com_t/com_f outputs are unused by the
// reference, so they are skipped. Dot products use b128 bf16 loads.
// ---------------------------------------------------------------------------
__global__ __launch_bounds__(128) void window_ca(
    const float* __restrict__ qc, const bf16* __restrict__ kl,
    const bf16* __restrict__ vl, const float* __restrict__ tcen,
    const float* __restrict__ fcen, float* __restrict__ outp)
{
  __shared__ int   flat[128];
  __shared__ float av[128];
  __shared__ float red[128];
  int s = blockIdx.x, b = blockIdx.y, w = threadIdx.x;

  float tcf = tcen[b * 1024 + s];
  float fcf = fcen[b * 1024 + s];
  int tc = min(max((int)roundf(tcf * 511.0f), 0), 511);
  int fc = min(max((int)roundf(fcf * 63.0f),  0), 63);
  int it = w >> 3, jf = w & 7;          // WT=16 x WF=8 window
  int tt = min(max(tc + it - 8, 0), 511);
  int ff = min(max(fc + jf - 4, 0), 63);
  flat[w] = b * 32768 + tt * 64 + ff;
  __syncthreads();

  const float* qrow = qc + ((size_t)(b * 1024 + s)) * 512;
  float* orow = outp + ((size_t)(b * 1024 + s)) * 512;

  for (int h = 0; h < 8; ++h) {
    const v8bf* kr8 = (const v8bf*)(kl + (size_t)flat[w] * 512 + h * 64);
    float scv = 0.0f;
#pragma unroll
    for (int c8 = 0; c8 < 8; ++c8) {
      v8bf kv = kr8[c8];
#pragma unroll
      for (int i = 0; i < 8; ++i) scv += qrow[h * 64 + c8 * 8 + i] * (float)kv[i];
    }
    scv *= 0.125f;

    red[w] = scv; __syncthreads();
    for (int o = 64; o > 0; o >>= 1) { if (w < o) red[w] = fmaxf(red[w], red[w + o]); __syncthreads(); }
    float mx = red[0]; __syncthreads();
    float e = __expf(scv - mx);
    av[w] = e; red[w] = e; __syncthreads();
    for (int o = 64; o > 0; o >>= 1) { if (w < o) red[w] += red[w + o]; __syncthreads(); }
    float sminv = 1.0f / red[0]; __syncthreads();

    if (w < 64) {
      float acc = 0.0f;
      for (int j = 0; j < 128; ++j)
        acc += av[j] * (float)vl[(size_t)flat[j] * 512 + h * 64 + w];
      orow[h * 64 + w] = acc * sminv;
    }
    __syncthreads();
  }
}

// ---------------------------------------------------------------------------
extern "C" void kernel_launch(void* const* d_in, const int* in_sizes, int n_in,
                              void* d_out, int out_size, void* d_ws, size_t ws_size,
                              hipStream_t stream) {
  const float* tgt    = (const float*)d_in[0];
  const float* memory = (const float*)d_in[1];
  // d_in[2] spatial_shapes, d_in[3] level_start_index, d_in[4] valid_ratios:
  // constants for ACTIVE=(0,) (T=512, F=64, start=0), folded into the kernels.
  const float* tcen = (const float*)d_in[5];
  const float* fcen = (const float*)d_in[6];
  const float* w_qkv = (const float*)d_in[7];
  const float* b_qkv = (const float*)d_in[8];
  const float* w_o = (const float*)d_in[9];
  const float* b_o = (const float*)d_in[10];
  const float* g1 = (const float*)d_in[11];
  const float* be1 = (const float*)d_in[12];
  const float* w_q = (const float*)d_in[13];
  const float* b_q = (const float*)d_in[14];
  const float* w_k = (const float*)d_in[15];
  const float* b_k = (const float*)d_in[16];
  const float* w_v = (const float*)d_in[17];
  const float* b_v = (const float*)d_in[18];
  const float* w_wo = (const float*)d_in[19];
  const float* b_wo = (const float*)d_in[20];
  const float* w_r = (const float*)d_in[21];
  const float* b_r = (const float*)d_in[22];
  const float* ctmp = (const float*)d_in[23];
  const float* g2 = (const float*)d_in[24];
  const float* be2 = (const float*)d_in[25];
  const float* w_f1 = (const float*)d_in[26];
  const float* b_f1 = (const float*)d_in[27];
  const float* w_f2 = (const float*)d_in[28];
  const float* b_f2 = (const float*)d_in[29];
  const float* g3 = (const float*)d_in[30];
  const float* be3 = (const float*)d_in[31];
  float* out = (float*)d_out;

  // Workspace carving (~201 MB total).
  float* qkvb    = (float*)d_ws;           // [2048,1536]
  float* sa_attn = qkvb + 3145728;         // [2048,512]
  float* sa_proj = sa_attn + 1048576;      // [2048,512]
  float* sa_nrm  = sa_proj + 1048576;      // [2048,512]
  float* qc      = sa_nrm + 1048576;       // [2048,512]
  float* sar     = qc + 1048576;           // [2048,512]
  float* ca_pre  = sar + 1048576;          // [2048,512]
  float* ca_out  = ca_pre + 1048576;       // [2048,512]
  float* tmid    = ca_out + 1048576;       // [2048,512]
  float* ffh     = tmid + 1048576;         // [2048,2048]
  float* ffo     = ffh + 4194304;          // [2048,512]
  bf16*  klv     = (bf16*)(ffo + 1048576); // [2,32768,512] bf16
  bf16*  vlv     = klv + 33554432;         // [2,32768,512] bf16

  dim3 blk(256);

  // 1. QKV projection.
  gemm_bf16_wmma<<<dim3(24, 64), blk, 0, stream>>>(
      tgt, w_qkv, b_qkv, nullptr, qkvb, nullptr, 2048, 1536, 512, 0);
  // 2. Causal self-attention (RoPE fused).
  flash_sa_rope<<<dim3(64, 8, 2), blk, 0, stream>>>(qkvb, sa_attn);
  // 3. Output projection + residual.
  gemm_bf16_wmma<<<dim3(8, 64), blk, 0, stream>>>(
      sa_attn, w_o, b_o, tgt, sa_proj, nullptr, 2048, 512, 512, 0);
  // 4. LN1.
  ln_kernel<<<dim3(2048), blk, 0, stream>>>(sa_proj, nullptr, g1, be1, sa_nrm);
  // 5. qc and sa_r projections.
  gemm_bf16_wmma<<<dim3(8, 64), blk, 0, stream>>>(
      sa_nrm, w_q, b_q, nullptr, qc, nullptr, 2048, 512, 512, 0);
  gemm_bf16_wmma<<<dim3(8, 64), blk, 0, stream>>>(
      sa_nrm, w_r, b_r, nullptr, sar, nullptr, 2048, 512, 512, 0);
  // 6. Level-0 K/V projections (only the active level), bf16 outputs.
  for (int b = 0; b < 2; ++b) {
    const float* memb = memory + (size_t)b * 43680 * 512;
    gemm_bf16_wmma<<<dim3(8, 1024), blk, 0, stream>>>(
        memb, w_k, b_k, nullptr, nullptr, klv + (size_t)b * 32768 * 512,
        32768, 512, 512, 0);
    gemm_bf16_wmma<<<dim3(8, 1024), blk, 0, stream>>>(
        memb, w_v, b_v, nullptr, nullptr, vlv + (size_t)b * 32768 * 512,
        32768, 512, 512, 0);
  }
  // 7. Window cross-attention.
  window_ca<<<dim3(1024, 2), dim3(128), 0, stream>>>(qc, klv, vlv, tcen, fcen, ca_pre);
  // 8. w_wo projection.
  gemm_bf16_wmma<<<dim3(8, 64), blk, 0, stream>>>(
      ca_pre, w_wo, b_wo, nullptr, ca_out, nullptr, 2048, 512, 512, 0);
  // 9. Ducking gate + LN2.
  gate_ln<<<dim3(2048), blk, 0, stream>>>(sar, ca_out, ctmp, g2, be2, tmid);
  // 10. FFN (GELU fused into first GEMM, residual into second).
  gemm_bf16_wmma<<<dim3(32, 64), blk, 0, stream>>>(
      tmid, w_f1, b_f1, nullptr, ffh, nullptr, 2048, 2048, 512, 1);
  gemm_bf16_wmma<<<dim3(8, 64), blk, 0, stream>>>(
      ffh, w_f2, b_f2, tmid, ffo, nullptr, 2048, 512, 2048, 0);
  // 11. LN3 -> output.
  ln_kernel<<<dim3(2048), blk, 0, stream>>>(ffo, nullptr, g3, be3, out);
}
